// RNNDecoder_8830452761314
// MI455X (gfx1250) — compile-verified, matrix operand order checked by reading, not run on hardware
//
#include <hip/hip_runtime.h>

typedef __attribute__((ext_vector_type(16))) __bf16 v16bf;
typedef __attribute__((ext_vector_type(8)))  float  v8f;

#define B_N   4096
#define L_N   512
#define H_N   2048
#define G3    1536
#define STEPS 32
#define LDS_LDA 520   // padded row stride (bf16 elems) -> conflict-free ds_load_b128

typedef __attribute__((address_space(3))) unsigned short lds_ushort;

// plain builtin 16B vectors (no C++ class => AS-qualified loads are legal)
typedef unsigned int v4u __attribute__((vector_size(16)));
typedef __attribute__((address_space(3))) const v4u lds_cv4u;

// exact pointee shape the async-LDS builtin wants: 16B int vector
typedef int v4i_t __attribute__((vector_size(16)));
typedef __attribute__((address_space(1))) v4i_t g_v4i;
typedef __attribute__((address_space(3))) v4i_t l_v4i;

// ---------- helpers ----------

__device__ __forceinline__ unsigned short f2bf(float x) {
  unsigned int u = __float_as_uint(x);
  unsigned int r = (u + 0x7FFFu + ((u >> 16) & 1u)) >> 16;  // RNE
  return (unsigned short)r;
}

// 16-bit WMMA A/B fragment from GLOBAL: two contiguous 16B chunks per lane.
__device__ __forceinline__ v16bf load_frag(const unsigned short* p) {
  union { v16bf v; v4u q[2]; } f;
  f.q[0] = *(const v4u*)(p);        // K = khalf + 0..7
  f.q[1] = *(const v4u*)(p + 16);   // K = khalf + 16..23
  return f.v;
}

// Same fragment but from LDS (forces ds_load_b128).
__device__ __forceinline__ v16bf load_frag_lds(const unsigned short* p) {
  const lds_cv4u* lp = (const lds_cv4u*)p;
  union { v16bf v; v4u q[2]; } f;
  f.q[0] = lp[0];
  f.q[1] = lp[2];   // +32 bytes
  return f.v;
}

__device__ __forceinline__ v8f wmma_bf16(v16bf a, v16bf b, v8f c) {
  return __builtin_amdgcn_wmma_f32_16x16x32_bf16(false, a, false, b,
                                                 (short)0, c, false, false);
}

__device__ __forceinline__ void async_copy_b128(const unsigned short* gsrc,
                                                unsigned short* ldst) {
#if __has_builtin(__builtin_amdgcn_global_load_async_to_lds_b128)
  __builtin_amdgcn_global_load_async_to_lds_b128(
      (g_v4i*)(unsigned short*)gsrc, (l_v4i*)ldst, 0, 0);
#else
  unsigned loff = (unsigned)(size_t)(lds_ushort*)ldst;
  asm volatile("global_load_async_to_lds_b128 %0, %1, off"
               :: "v"(loff), "v"(gsrc) : "memory");
#endif
}

__device__ __forceinline__ void wait_async0() {
#if __has_builtin(__builtin_amdgcn_s_wait_asynccnt)
  __builtin_amdgcn_s_wait_asynccnt(0);
#else
  asm volatile("s_wait_asynccnt 0x0" ::: "memory");
#endif
}

// Stage 16 rows x 512 bf16 (row-major, ld=512) into sA with padded stride.
// 1024 x 16B chunks, 4 per thread, via async global->LDS copies.
__device__ __forceinline__ void stage_tile(const unsigned short* gbase,
                                           unsigned short* sA) {
#pragma unroll
  for (int i = 0; i < 4; ++i) {
    int chunk = threadIdx.x + i * 256;
    int row = chunk >> 6, cin = chunk & 63;
    async_copy_b128(gbase + (size_t)row * L_N + cin * 8,
                    sA + row * LDS_LDA + cin * 8);
  }
  wait_async0();       // my async writes to LDS have landed
  __syncthreads();     // everyone's have
}

__device__ __forceinline__ float sigmoid_fast(float x) {
  return 1.0f / (1.0f + __expf(-x));
}
__device__ __forceinline__ float tanh_fast(float x) {
  float e = __expf(-2.0f * x);
  return (1.0f - e) / (1.0f + e);
}
__device__ __forceinline__ float leaky(float x) {
  return x >= 0.0f ? x : 0.01f * x;
}

// ---------- setup kernels ----------

__global__ void cvt_bf16_kernel(const float* __restrict__ src,
                                unsigned short* __restrict__ dst, int n) {
  int i = blockIdx.x * blockDim.x + threadIdx.x;
  if (i < n) dst[i] = f2bf(src[i]);
}

__global__ void zero_h_kernel(float* __restrict__ h32,
                              unsigned short* __restrict__ h16) {
  int i = blockIdx.x * blockDim.x + threadIdx.x;
  if (i < B_N * L_N) { h32[i] = 0.0f; h16[i] = 0; }
}

// ---------- gi = z @ W_ih^T + b_ih  (B x 1536, K=512) ----------
// block: 16 rows x 1536 cols; 8 waves x 12 col-tiles; B double-buffered in
// groups of 4 tiles; A from LDS.

__global__ void __launch_bounds__(256)
gi_kernel(const unsigned short* __restrict__ zbf,
          const unsigned short* __restrict__ wihbf,
          const float* __restrict__ b_ih,
          float* __restrict__ gi) {
  __shared__ unsigned short sA[16 * LDS_LDA];
  const int lane  = threadIdx.x & 31;
  const int wave  = threadIdx.x >> 5;
  const int m0    = blockIdx.x * 16;
  const int khalf = (lane >> 4) * 8;
  const int ncol  = wave * 192 + (lane & 15);

  stage_tile(zbf + (size_t)m0 * L_N, sA);

  v8f acc[12] = {};
  v16bf bbuf[2][4];
  const unsigned short* sArow = sA + (lane & 15) * LDS_LDA + khalf;

#define GI_LOADB(buf, kt, g)                                                   \
  {                                                                            \
    _Pragma("unroll") for (int t = 0; t < 4; ++t) {                            \
      int n = ncol + ((g) * 4 + t) * 16;                                       \
      bbuf[buf][t] =                                                           \
          load_frag(wihbf + (size_t)n * L_N + (kt) * 32 + khalf);              \
    }                                                                          \
  }

  GI_LOADB(0, 0, 0);
#pragma unroll
  for (int kt = 0; kt < 16; ++kt) {
    v16bf a = load_frag_lds(sArow + kt * 32);
#pragma unroll
    for (int g = 0; g < 3; ++g) {
      const int cur = (kt * 3 + g) & 1;
      if (!(kt == 15 && g == 2)) {
        int nkt = (g == 2) ? kt + 1 : kt;
        int ng  = (g == 2) ? 0 : g + 1;
        GI_LOADB(cur ^ 1, nkt, ng);
      }
#pragma unroll
      for (int t = 0; t < 4; ++t)
        acc[g * 4 + t] = wmma_bf16(a, bbuf[cur][t], acc[g * 4 + t]);
    }
  }
#undef GI_LOADB

  const int rbase = m0 + ((lane >> 4) << 3);
#pragma unroll
  for (int c = 0; c < 12; ++c) {
    int col = ncol + c * 16;
    float bias = b_ih[col];
#pragma unroll
    for (int r = 0; r < 8; ++r)
      gi[(size_t)(rbase + r) * G3 + col] = acc[c][r] + bias;
  }
}

// ---------- one GRU step: h' from h (in-place) ----------
// wave owns matched (r,u,n) column triples -> gate math in registers.
// B prefetch groups = the 3 fragments of one column tile.

__global__ void __launch_bounds__(256)
gru_step_kernel(const unsigned short* __restrict__ whhbf,
                const float* __restrict__ b_hh,
                const float* __restrict__ gi,
                float* __restrict__ h32,
                unsigned short* __restrict__ h16) {
  __shared__ unsigned short sA[16 * LDS_LDA];
  const int lane  = threadIdx.x & 31;
  const int wave  = threadIdx.x >> 5;
  const int m0    = blockIdx.x * 16;
  const int khalf = (lane >> 4) * 8;
  const int ncol  = wave * 64 + (lane & 15);

  stage_tile(h16 + (size_t)m0 * L_N, sA);   // all global h16 reads end here

  v8f accR[4] = {}, accU[4] = {}, accN[4] = {};
  v16bf bbuf[2][3];
  const unsigned short* sArow = sA + (lane & 15) * LDS_LDA + khalf;

#define GS_LOADB(buf, kt, c)                                                   \
  {                                                                            \
    int n = ncol + (c) * 16;                                                   \
    bbuf[buf][0] = load_frag(whhbf + (size_t)n * L_N + (kt) * 32 + khalf);     \
    bbuf[buf][1] =                                                             \
        load_frag(whhbf + (size_t)(512 + n) * L_N + (kt) * 32 + khalf);        \
    bbuf[buf][2] =                                                             \
        load_frag(whhbf + (size_t)(1024 + n) * L_N + (kt) * 32 + khalf);       \
  }

  GS_LOADB(0, 0, 0);
#pragma unroll
  for (int kt = 0; kt < 16; ++kt) {
    v16bf a = load_frag_lds(sArow + kt * 32);
#pragma unroll
    for (int c = 0; c < 4; ++c) {
      const int cur = c & 1;   // 4 groups/kt -> parity resets each kt
      if (!(kt == 15 && c == 3)) {
        int nkt = (c == 3) ? kt + 1 : kt;
        int nc  = (c + 1) & 3;
        GS_LOADB(cur ^ 1, nkt, nc);
      }
      accR[c] = wmma_bf16(a, bbuf[cur][0], accR[c]);
      accU[c] = wmma_bf16(a, bbuf[cur][1], accU[c]);
      accN[c] = wmma_bf16(a, bbuf[cur][2], accN[c]);
    }
  }
#undef GS_LOADB

  const int rbase = m0 + ((lane >> 4) << 3);
#pragma unroll
  for (int c = 0; c < 4; ++c) {
    int col = ncol + c * 16;
    float bhr = b_hh[col], bhu = b_hh[512 + col], bhn = b_hh[1024 + col];
#pragma unroll
    for (int r = 0; r < 8; ++r) {
      int row = rbase + r;
      float gr = accR[c][r] + bhr + gi[(size_t)row * G3 + col];
      float gu = accU[c][r] + bhu + gi[(size_t)row * G3 + 512 + col];
      float hn = accN[c][r] + bhn;
      float gn = gi[(size_t)row * G3 + 1024 + col];
      float rg = sigmoid_fast(gr);
      float ug = sigmoid_fast(gu);
      float ng = tanh_fast(gn + rg * hn);
      float hp = h32[(size_t)row * L_N + col];
      float hv = (1.0f - ug) * ng + ug * hp;
      h32[(size_t)row * L_N + col] = hv;
      h16[(size_t)row * L_N + col] = f2bf(hv);
    }
  }
}

// ---------- hid = leaky(h@W1^T+b1); o = leaky(hid@W2^T+b2) fused ----------

__global__ void __launch_bounds__(256)
decode_out_kernel(const unsigned short* __restrict__ w1bf,
                  const float* __restrict__ b1,
                  const float* __restrict__ w2,
                  const float* __restrict__ b2,
                  const unsigned short* __restrict__ h16,
                  float* __restrict__ out_step) {
  __shared__ unsigned short sA[16 * LDS_LDA];
  __shared__ float sred[8][2][8][4];
  const int lane  = threadIdx.x & 31;
  const int wave  = threadIdx.x >> 5;
  const int m0    = blockIdx.x * 16;
  const int khalf = (lane >> 4) * 8;
  const int ncol  = wave * 256 + (lane & 15);

  stage_tile(h16 + (size_t)m0 * L_N, sA);

  v8f acc[16] = {};
  v16bf bbuf[2][4];
  const unsigned short* sArow = sA + (lane & 15) * LDS_LDA + khalf;

#define DO_LOADB(buf, kt, g)                                                   \
  {                                                                            \
    _Pragma("unroll") for (int t = 0; t < 4; ++t) {                            \
      int n = ncol + ((g) * 4 + t) * 16;                                       \
      bbuf[buf][t] =                                                           \
          load_frag(w1bf + (size_t)n * L_N + (kt) * 32 + khalf);               \
    }                                                                          \
  }

  DO_LOADB(0, 0, 0);
#pragma unroll
  for (int kt = 0; kt < 16; ++kt) {
    v16bf a = load_frag_lds(sArow + kt * 32);
#pragma unroll
    for (int g = 0; g < 4; ++g) {
      const int cur = g & 1;
      if (!(kt == 15 && g == 3)) {
        int nkt = (g == 3) ? kt + 1 : kt;
        int ng  = (g + 1) & 3;
        DO_LOADB(cur ^ 1, nkt, ng);
      }
#pragma unroll
      for (int t = 0; t < 4; ++t)
        acc[g * 4 + t] = wmma_bf16(a, bbuf[cur][t], acc[g * 4 + t]);
    }
  }
#undef DO_LOADB

  float part[8][4] = {};
#pragma unroll
  for (int c = 0; c < 16; ++c) {
    int col = ncol + c * 16;
    float bb  = b1[col];
    float w20 = w2[col],           w21 = w2[H_N + col];
    float w22 = w2[2 * H_N + col], w23 = w2[3 * H_N + col];
#pragma unroll
    for (int r = 0; r < 8; ++r) {
      float v = leaky(acc[c][r] + bb);
      part[r][0] += v * w20; part[r][1] += v * w21;
      part[r][2] += v * w22; part[r][3] += v * w23;
    }
  }
#pragma unroll
  for (int off = 1; off < 16; off <<= 1)
#pragma unroll
    for (int r = 0; r < 8; ++r)
#pragma unroll
      for (int j = 0; j < 4; ++j)
        part[r][j] += __shfl_xor(part[r][j], off, 32);

  if ((lane & 15) == 0) {
    int half = lane >> 4;
#pragma unroll
    for (int r = 0; r < 8; ++r)
#pragma unroll
      for (int j = 0; j < 4; ++j)
        sred[wave][half][r][j] = part[r][j];
  }
  __syncthreads();

  int t = threadIdx.x;
  if (t < 64) {
    int rl = t & 15, j = t >> 4;
    float s = 0.0f;
#pragma unroll
    for (int w = 0; w < 8; ++w) s += sred[w][rl >> 3][rl & 7][j];
    s = leaky(s + b2[j]);
    out_step[(size_t)(m0 + rl) * 4 + j] = s;
  }
}

// ---------- host ----------

extern "C" void kernel_launch(void* const* d_in, const int* in_sizes, int n_in,
                              void* d_out, int out_size, void* d_ws, size_t ws_size,
                              hipStream_t stream) {
  const float* z    = (const float*)d_in[0];
  const float* W_ih = (const float*)d_in[1];
  const float* W_hh = (const float*)d_in[2];
  const float* b_ih = (const float*)d_in[3];
  const float* b_hh = (const float*)d_in[4];
  const float* W1   = (const float*)d_in[5];
  const float* b1   = (const float*)d_in[6];
  const float* W2   = (const float*)d_in[7];
  const float* b2   = (const float*)d_in[8];
  // d_in[9] = n_steps (device scalar); loop trip count must be host-side => 32.
  float* out = (float*)d_out;

  char* ws = (char*)d_ws;
  unsigned short* whhbf = (unsigned short*)ws; ws += (size_t)G3 * L_N * 2;
  unsigned short* wihbf = (unsigned short*)ws; ws += (size_t)G3 * L_N * 2;
  unsigned short* w1bf  = (unsigned short*)ws; ws += (size_t)H_N * L_N * 2;
  unsigned short* zbf   = (unsigned short*)ws; ws += (size_t)B_N * L_N * 2;
  float*          gi    = (float*)ws;          ws += (size_t)B_N * G3 * 4;
  float*          h32   = (float*)ws;          ws += (size_t)B_N * L_N * 4;
  unsigned short* h16   = (unsigned short*)ws;

  cvt_bf16_kernel<<<(G3 * L_N + 255) / 256, 256, 0, stream>>>(W_hh, whhbf, G3 * L_N);
  cvt_bf16_kernel<<<(G3 * L_N + 255) / 256, 256, 0, stream>>>(W_ih, wihbf, G3 * L_N);
  cvt_bf16_kernel<<<(H_N * L_N + 255) / 256, 256, 0, stream>>>(W1, w1bf, H_N * L_N);
  cvt_bf16_kernel<<<(B_N * L_N + 255) / 256, 256, 0, stream>>>(z, zbf, B_N * L_N);
  zero_h_kernel<<<(B_N * L_N + 255) / 256, 256, 0, stream>>>(h32, h16);

  gi_kernel<<<B_N / 16, 256, 0, stream>>>(zbf, wihbf, b_ih, gi);

  for (int s = 0; s < STEPS; ++s) {
    gru_step_kernel<<<B_N / 16, 256, 0, stream>>>(whhbf, b_hh, gi, h32, h16);
    decode_out_kernel<<<B_N / 16, 256, 0, stream>>>(w1bf, b1, W2, b2, h16,
                                                    out + (size_t)s * B_N * 4);
  }
}
